// TransformerEncoderLayer_69157563400510
// MI455X (gfx1250) — compile-verified, matrix operand order checked by reading, not run on hardware
//
#include <hip/hip_runtime.h>
#include <math.h>

// ---------------------------------------------------------------------------
// Types
// ---------------------------------------------------------------------------
typedef __bf16 v16bf __attribute__((ext_vector_type(16)));
typedef __bf16 v8bf  __attribute__((ext_vector_type(8)));
typedef float  v8f   __attribute__((ext_vector_type(8)));

static __device__ __forceinline__ unsigned short f2bfbits(float x) {
    unsigned u = __builtin_bit_cast(unsigned, x);
    unsigned r = u + 0x7fffu + ((u >> 16) & 1u);   // round-to-nearest-even
    return (unsigned short)(r >> 16);
}

static __device__ __forceinline__ float geluf(float v) {
    return 0.5f * v * (1.0f + erff(v * 0.70710678118654752f));
}

// Build a 16-element bf16 WMMA fragment from two 16-byte-aligned runs of
// 8 bf16 (K offsets 8g and 8g+16 per the CDNA5 16-bit A/B layout).
static __device__ __forceinline__ v16bf make_frag(const unsigned short* lo,
                                                  const unsigned short* hi) {
    v8bf a = *reinterpret_cast<const v8bf*>(lo);
    v8bf b = *reinterpret_cast<const v8bf*>(hi);
    return __builtin_shufflevector(a, b, 0, 1, 2, 3, 4, 5, 6, 7,
                                         8, 9, 10, 11, 12, 13, 14, 15);
}

#define WMMA_BF16(A, B, C) \
    __builtin_amdgcn_wmma_f32_16x16x32_bf16(false, (A), false, (B), (short)0, (C), false, false)

// ---------------------------------------------------------------------------
// Weight conversion kernels (fp32 -> bf16)
// ---------------------------------------------------------------------------
__global__ void cvt_plain_kernel(const float* __restrict__ in,
                                 unsigned short* __restrict__ out, int n) {
    int i = blockIdx.x * 256 + threadIdx.x;
    if (i < n) out[i] = f2bfbits(in[i]);
}

// (H, D, HD) -> column-combined (D, H*HD): out[d*H*HD + h*HD + e] = in[h,d,e]
__global__ void cvt_qkvw_kernel(const float* __restrict__ in,
                                unsigned short* __restrict__ out,
                                int H, int D, int HD) {
    int i = blockIdx.x * 256 + threadIdx.x;
    int n = H * D * HD;
    if (i >= n) return;
    int h = i / (D * HD);
    int r = i - h * (D * HD);
    int d = r / HD;
    int e = r - d * HD;
    out[(size_t)d * (H * HD) + h * HD + e] = f2bfbits(in[i]);
}

// ---------------------------------------------------------------------------
// LayerNorm: one 256-thread block per token, bf16 output
// ---------------------------------------------------------------------------
__global__ __launch_bounds__(256) void ln_kernel(const float* __restrict__ x,
                                                 const float* __restrict__ g,
                                                 const float* __restrict__ b,
                                                 unsigned short* __restrict__ out,
                                                 int D) {
    const int row = blockIdx.x;
    const int tid = threadIdx.x;
    const float* xr = x + (size_t)row * D;
    __shared__ float red[20];

    float s = 0.f, ss = 0.f;
    for (int i = tid; i < D; i += 256) {
        float v = xr[i];
        s += v;
        ss += v * v;
    }
#pragma unroll
    for (int m = 16; m >= 1; m >>= 1) {
        s  += __shfl_xor(s,  m, 32);
        ss += __shfl_xor(ss, m, 32);
    }
    const int wid = tid >> 5;
    if ((tid & 31) == 0) { red[wid] = s; red[8 + wid] = ss; }
    __syncthreads();
    if (tid == 0) {
        float S = 0.f, SS = 0.f;
        for (int w = 0; w < 8; ++w) { S += red[w]; SS += red[8 + w]; }
        float mu  = S / (float)D;
        float var = SS / (float)D - mu * mu;
        red[16] = mu;
        red[17] = rsqrtf(var + 1e-5f);
    }
    __syncthreads();
    const float mu = red[16], rs = red[17];
    unsigned short* orow = out + (size_t)row * D;
    for (int i = tid; i < D; i += 256)
        orow[i] = f2bfbits((xr[i] - mu) * rs * g[i] + b[i]);
}

// ---------------------------------------------------------------------------
// Tiled bf16 WMMA GEMM (double-buffered, BK=64):
//   C[M,N] = [res +] act((A[M,K] @ B[K,N] + bias) * scale)
//   block tile 128x128, 8 waves (2 x 4), each wave: 64x32 via 4x2 fragments.
//   Pipeline: at iteration k, stage tile k+1 into the other LDS buffer, then
//   run the 16 WMMAs for tile k. One barrier per 64-wide K block.
// ---------------------------------------------------------------------------
constexpr int GL = 72;   // padded LDS row length (elements) for BK=64

template <int ACT, bool RES, bool OUTF, bool OUTB>
__global__ __launch_bounds__(256) void gemm_bf16_kernel(
        const unsigned short* __restrict__ A,   // [M,K] bf16 row-major
        const unsigned short* __restrict__ Bw,  // [K,N] bf16 row-major
        const float* __restrict__ bias,         // [N]
        const float* __restrict__ res,          // [M,N] fp32 (if RES)
        float* __restrict__ outF,               // [M,N] fp32 (if OUTF)
        unsigned short* __restrict__ outB,      // [M,N] bf16 (if OUTB)
        int M, int N, int K, float scale) {
    __shared__ alignas(16) unsigned short As [2 * 128 * GL];  // [buf][m][k]
    __shared__ alignas(16) unsigned short Bts[2 * 128 * GL];  // [buf][n][k]

    const int tid  = threadIdx.x;
    const int wid  = tid >> 5;
    const int lane = tid & 31;
    const int g    = lane >> 4;
    const int ln   = lane & 15;
    const int wm   = wid >> 2;
    const int wn   = wid & 3;
    const int m0   = blockIdx.y * 128;
    const int n0   = blockIdx.x * 128;

    // Stage a 128x64 A tile and a (transposed) 64x128 B tile into buffer `buf`.
    auto stageAB = [&](int buf, int k0) {
        unsigned short* as = &As [buf * 128 * GL];
        unsigned short* bt = &Bts[buf * 128 * GL];
#pragma unroll
        for (int i = 0; i < 4; ++i) {               // 1024 16B chunks of A
            int idx = tid + i * 256;
            int r   = idx >> 3;
            int kc  = (idx & 7) << 3;
            *reinterpret_cast<uint4*>(&as[r * GL + kc]) =
                *reinterpret_cast<const uint4*>(&A[(size_t)(m0 + r) * K + k0 + kc]);
        }
#pragma unroll
        for (int i = 0; i < 4; ++i) {               // 1024 8-wide B chunks
            int idx = tid + i * 256;
            int kk  = idx >> 4;
            int nc  = (idx & 15) << 3;
            uint4 d = *reinterpret_cast<const uint4*>(&Bw[(size_t)(k0 + kk) * N + n0 + nc]);
            const unsigned short* sp = reinterpret_cast<const unsigned short*>(&d);
#pragma unroll
            for (int j = 0; j < 8; ++j) bt[(nc + j) * GL + kk] = sp[j];
        }
    };

    v8f acc[4][2];
#pragma unroll
    for (int mf = 0; mf < 4; ++mf)
#pragma unroll
        for (int nf = 0; nf < 2; ++nf)
#pragma unroll
            for (int r = 0; r < 8; ++r) acc[mf][nf][r] = 0.f;

    const int KB = K >> 6;
    stageAB(0, 0);
    for (int kb = 0; kb < KB; ++kb) {
        __syncthreads();                 // stage(kb) visible; buf (kb+1)&1 free
        if (kb + 1 < KB) stageAB((kb + 1) & 1, (kb + 1) << 6);

        const unsigned short* as = &As [(kb & 1) * 128 * GL];
        const unsigned short* bt = &Bts[(kb & 1) * 128 * GL];
#pragma unroll
        for (int kd = 0; kd < 2; ++kd) {            // two 32-wide WMMA K-steps
            v16bf af[4];
#pragma unroll
            for (int mf = 0; mf < 4; ++mf) {
                const unsigned short* p =
                    &as[(wm * 64 + mf * 16 + ln) * GL + kd * 32 + 8 * g];
                af[mf] = make_frag(p, p + 16);
            }
#pragma unroll
            for (int nf = 0; nf < 2; ++nf) {
                const unsigned short* p =
                    &bt[(wn * 32 + nf * 16 + ln) * GL + kd * 32 + 8 * g];
                v16bf bf = make_frag(p, p + 16);
#pragma unroll
                for (int mf = 0; mf < 4; ++mf)
                    acc[mf][nf] = WMMA_BF16(af[mf], bf, acc[mf][nf]);
            }
        }
    }

    // ---- straight-line epilogue (all flags compile-time) ----
#pragma unroll
    for (int mf = 0; mf < 4; ++mf) {
#pragma unroll
        for (int nf = 0; nf < 2; ++nf) {
            int col  = n0 + wn * 32 + nf * 16 + ln;
            float bv = bias[col];
#pragma unroll
            for (int r = 0; r < 8; ++r) {
                int row = m0 + wm * 64 + mf * 16 + 8 * g + r;
                float v = (acc[mf][nf][r] + bv) * scale;
                if (ACT == 1) v = geluf(v);
                size_t o = (size_t)row * N + col;
                if (RES)  v += res[o];
                if (OUTF) outF[o] = v;
                if (OUTB) outB[o] = f2bfbits(v);
            }
        }
    }
}

// ---------------------------------------------------------------------------
// Causal flash attention (bf16 in/out, online softmax, WMMA for QK^T and PV).
//   Q/K/V layout: [B*S, D] bf16 with head column offset h*HD, row stride D.
//   grid = (S/128, B*H); block = 256 (8 waves); wave w owns q rows w*16..+15.
//   K/V tiles double-buffered in LDS.
// ---------------------------------------------------------------------------
__global__ __launch_bounds__(256) void attn_kernel(
        const unsigned short* __restrict__ Q,
        const unsigned short* __restrict__ Kb,
        const unsigned short* __restrict__ Vb,
        unsigned short* __restrict__ O,
        int S, int H, int HD) {
    const int D   = H * HD;                       // 1024
    const int qt  = blockIdx.x;
    const int bh  = blockIdx.y;
    const int b   = bh / H;
    const int h   = bh - b * H;
    const size_t base = (size_t)b * S * D + (size_t)h * HD;

    const int tid  = threadIdx.x;
    const int wid  = tid >> 5;
    const int lane = tid & 31;
    const int g    = lane >> 4;
    const int ln   = lane & 15;
    const int q0   = qt * 128;

    __shared__ alignas(16) unsigned short Ks [2 * 64 * GL];   // [buf][kc][d]
    __shared__ alignas(16) unsigned short Vts[2 * 64 * GL];   // [buf][d][kc]
    __shared__ alignas(16) unsigned short Ps [8 * 16 * GL];   // per-wave P

    auto stageKV = [&](int buf, int k0) {
        unsigned short* ks = &Ks [buf * 64 * GL];
        unsigned short* vt = &Vts[buf * 64 * GL];
#pragma unroll
        for (int i = 0; i < 2; ++i) {              // 512 16B chunks of K
            int idx = tid + i * 256;
            int kc  = idx >> 3;
            int dc  = (idx & 7) << 3;
            *reinterpret_cast<uint4*>(&ks[kc * GL + dc]) =
                *reinterpret_cast<const uint4*>(&Kb[base + (size_t)(k0 + kc) * D + dc]);
        }
#pragma unroll
        for (int i = 0; i < 2; ++i) {              // V transposed scatter
            int idx = tid + i * 256;
            int kc  = idx >> 3;
            int dc  = (idx & 7) << 3;
            uint4 d = *reinterpret_cast<const uint4*>(&Vb[base + (size_t)(k0 + kc) * D + dc]);
            const unsigned short* sp = reinterpret_cast<const unsigned short*>(&d);
#pragma unroll
            for (int j = 0; j < 8; ++j) vt[(dc + j) * GL + kc] = sp[j];
        }
    };

    // Q fragments, resident for the whole block (d split into 2 k-steps)
    const int qrow = q0 + wid * 16 + ln;
    v16bf qa[2];
#pragma unroll
    for (int kd = 0; kd < 2; ++kd) {
        const unsigned short* qp = &Q[base + (size_t)qrow * D + kd * 32 + 8 * g];
        qa[kd] = make_frag(qp, qp + 16);
    }

    v8f of[4];
#pragma unroll
    for (int j = 0; j < 4; ++j)
#pragma unroll
        for (int r = 0; r < 8; ++r) of[j][r] = 0.f;
    float mrow[8], lrow[8];
#pragma unroll
    for (int r = 0; r < 8; ++r) { mrow[r] = -1e30f; lrow[r] = 0.f; }

    const int kend = q0 + 128;                    // causal: cols <= q0+127
    stageKV(0, 0);
    int it = 0;
    for (int k0 = 0; k0 < kend; k0 += 64, ++it) {
        __syncthreads();
        if (k0 + 64 < kend) stageKV((it + 1) & 1, k0 + 64);
        const unsigned short* ks = &Ks [(it & 1) * 64 * GL];
        const unsigned short* vt = &Vts[(it & 1) * 64 * GL];

        // ---- S = Q K^T  (16 x 64 per wave; 1/sqrt(HD) folded into Q) ----
        v8f sf[4];
#pragma unroll
        for (int nf = 0; nf < 4; ++nf) {
            v8f c;
#pragma unroll
            for (int r = 0; r < 8; ++r) c[r] = 0.f;
#pragma unroll
            for (int kd = 0; kd < 2; ++kd) {
                const unsigned short* kp = &ks[(nf * 16 + ln) * GL + kd * 32 + 8 * g];
                v16bf bf = make_frag(kp, kp + 16);
                c = WMMA_BF16(qa[kd], bf, c);
            }
            sf[nf] = c;
        }

        // ---- causal mask (uniform-per-wave predicate) ----
        if (k0 + 63 > q0 + wid * 16) {
#pragma unroll
            for (int nf = 0; nf < 4; ++nf) {
                int col = k0 + nf * 16 + ln;
#pragma unroll
                for (int r = 0; r < 8; ++r) {
                    int row = q0 + wid * 16 + 8 * g + r;
                    sf[nf][r] = (col > row) ? -1e30f : sf[nf][r];
                }
            }
        }

        // ---- online softmax (row = r + 8g, reduce across 16-lane half) ----
        // Arguments of the exps are always <= 0, so the fast __expf
        // (single v_exp_f32) is safe: it just underflows to 0.
        float alpha[8];
#pragma unroll
        for (int r = 0; r < 8; ++r) {
            float mx = fmaxf(fmaxf(sf[0][r], sf[1][r]), fmaxf(sf[2][r], sf[3][r]));
#pragma unroll
            for (int m = 8; m >= 1; m >>= 1) mx = fmaxf(mx, __shfl_xor(mx, m, 32));
            float mnew = fmaxf(mrow[r], mx);
            alpha[r] = __expf(mrow[r] - mnew);
            float ps = 0.f;
#pragma unroll
            for (int nf = 0; nf < 4; ++nf) {
                float p = __expf(sf[nf][r] - mnew);
                sf[nf][r] = p;
                ps += p;
            }
#pragma unroll
            for (int m = 8; m >= 1; m >>= 1) ps += __shfl_xor(ps, m, 32);
            lrow[r] = lrow[r] * alpha[r] + ps;
            mrow[r] = mnew;
        }

        // ---- write P to this wave's LDS slab (C-layout -> A-layout) ----
        unsigned short* pw = &Ps[wid * 16 * GL];
#pragma unroll
        for (int nf = 0; nf < 4; ++nf)
#pragma unroll
            for (int r = 0; r < 8; ++r)
                pw[(8 * g + r) * GL + nf * 16 + ln] = f2bfbits(sf[nf][r]);

        // ---- rescale O, then O += P @ V ----
#pragma unroll
        for (int j = 0; j < 4; ++j)
#pragma unroll
            for (int r = 0; r < 8; ++r) of[j][r] *= alpha[r];

        v16bf pa[2];
#pragma unroll
        for (int kd = 0; kd < 2; ++kd) {           // same-wave LDS RAW: in-order
            const unsigned short* pp = &pw[ln * GL + kd * 32 + 8 * g];
            pa[kd] = make_frag(pp, pp + 16);
        }
#pragma unroll
        for (int j = 0; j < 4; ++j) {
#pragma unroll
            for (int kd = 0; kd < 2; ++kd) {
                const unsigned short* vp = &vt[(j * 16 + ln) * GL + kd * 32 + 8 * g];
                v16bf vb = make_frag(vp, vp + 16);
                of[j] = WMMA_BF16(pa[kd], vb, of[j]);
            }
        }
    }

    // ---- normalize (8 reciprocals instead of 32 divides) and store bf16 ----
    float inv[8];
#pragma unroll
    for (int r = 0; r < 8; ++r) inv[r] = 1.0f / lrow[r];
#pragma unroll
    for (int j = 0; j < 4; ++j) {
#pragma unroll
        for (int r = 0; r < 8; ++r) {
            int row = q0 + wid * 16 + 8 * g + r;
            float v = of[j][r] * inv[r];
            O[base + (size_t)row * D + j * 16 + ln] = f2bfbits(v);
        }
    }
}

// ---------------------------------------------------------------------------
// Host orchestration
// ---------------------------------------------------------------------------
extern "C" void kernel_launch(void* const* d_in, const int* in_sizes, int n_in,
                              void* d_out, int out_size, void* d_ws, size_t ws_size,
                              hipStream_t stream) {
    constexpr int B = 4, S = 2048, D = 1024, H = 16, HD = 64, FF = 4096;
    constexpr int NT = B * S;                       // 8192 tokens

    const float* x     = (const float*)d_in[0];
    const float* Wq    = (const float*)d_in[1];
    const float* bq    = (const float*)d_in[2];
    const float* Wk    = (const float*)d_in[3];
    const float* bk    = (const float*)d_in[4];
    const float* Wv    = (const float*)d_in[5];
    const float* bv    = (const float*)d_in[6];
    const float* Wo    = (const float*)d_in[7];
    const float* bo    = (const float*)d_in[8];
    const float* W1    = (const float*)d_in[9];
    const float* b1    = (const float*)d_in[10];
    const float* W2    = (const float*)d_in[11];
    const float* b2    = (const float*)d_in[12];
    const float* ln1_g = (const float*)d_in[13];
    const float* ln1_b = (const float*)d_in[14];
    const float* ln2_g = (const float*)d_in[15];
    const float* ln2_b = (const float*)d_in[16];
    float* out = (float*)d_out;

    // ---- workspace carve-up (bf16 stored as unsigned short) ----
    char* p = (char*)d_ws;
    auto alloc = [&](size_t bytes) {
        void* r = p;
        p += (bytes + 255) & ~(size_t)255;
        return r;
    };
    unsigned short* wq_bf  = (unsigned short*)alloc((size_t)D * D * 2);
    unsigned short* wk_bf  = (unsigned short*)alloc((size_t)D * D * 2);
    unsigned short* wv_bf  = (unsigned short*)alloc((size_t)D * D * 2);
    unsigned short* wo_bf  = (unsigned short*)alloc((size_t)D * D * 2);
    unsigned short* w1_bf  = (unsigned short*)alloc((size_t)D * FF * 2);
    unsigned short* w2_bf  = (unsigned short*)alloc((size_t)FF * D * 2);
    unsigned short* h_bf   = (unsigned short*)alloc((size_t)NT * D * 2);   // h1 then h2
    unsigned short* q_bf   = (unsigned short*)alloc((size_t)NT * D * 2);
    unsigned short* k_bf   = (unsigned short*)alloc((size_t)NT * D * 2);
    unsigned short* v_bf   = (unsigned short*)alloc((size_t)NT * D * 2);
    unsigned short* at_bf  = (unsigned short*)alloc((size_t)NT * D * 2);
    float*          x2_f   = (float*)alloc((size_t)NT * D * 4);
    unsigned short* y1_bf  = (unsigned short*)alloc((size_t)NT * FF * 2);
    (void)ws_size; (void)in_sizes; (void)n_in; (void)out_size;

    // ---- weight conversion ----
    {
        int n = H * D * HD;
        int blk = (n + 255) / 256;
        cvt_qkvw_kernel<<<blk, 256, 0, stream>>>(Wq, wq_bf, H, D, HD);
        cvt_qkvw_kernel<<<blk, 256, 0, stream>>>(Wk, wk_bf, H, D, HD);
        cvt_qkvw_kernel<<<blk, 256, 0, stream>>>(Wv, wv_bf, H, D, HD);
        cvt_plain_kernel<<<(D * D + 255) / 256, 256, 0, stream>>>(Wo, wo_bf, D * D);
        cvt_plain_kernel<<<(D * FF + 255) / 256, 256, 0, stream>>>(W1, w1_bf, D * FF);
        cvt_plain_kernel<<<(FF * D + 255) / 256, 256, 0, stream>>>(W2, w2_bf, FF * D);
    }

    // ---- LN1 ----
    ln_kernel<<<NT, 256, 0, stream>>>(x, ln1_g, ln1_b, h_bf, D);

    // ---- QKV projections (Q pre-scaled by 1/sqrt(HD)) ----
    dim3 gP(D / 128, NT / 128);
    gemm_bf16_kernel<0, false, false, true><<<gP, 256, 0, stream>>>(
        h_bf, wq_bf, bq, nullptr, nullptr, q_bf, NT, D, D, 0.125f);
    gemm_bf16_kernel<0, false, false, true><<<gP, 256, 0, stream>>>(
        h_bf, wk_bf, bk, nullptr, nullptr, k_bf, NT, D, D, 1.0f);
    gemm_bf16_kernel<0, false, false, true><<<gP, 256, 0, stream>>>(
        h_bf, wv_bf, bv, nullptr, nullptr, v_bf, NT, D, D, 1.0f);

    // ---- causal flash attention ----
    attn_kernel<<<dim3(S / 128, B * H), 256, 0, stream>>>(q_bf, k_bf, v_bf, at_bf,
                                                          S, H, HD);

    // ---- output projection + residual -> x2 (fp32) ----
    gemm_bf16_kernel<0, true, true, false><<<gP, 256, 0, stream>>>(
        at_bf, wo_bf, bo, x, x2_f, nullptr, NT, D, D, 1.0f);

    // ---- LN2 ----
    ln_kernel<<<NT, 256, 0, stream>>>(x2_f, ln2_g, ln2_b, h_bf, D);

    // ---- FFN: gelu(h2 @ W1 + b1) -> y1 (bf16) ----
    dim3 gF(FF / 128, NT / 128);
    gemm_bf16_kernel<1, false, false, true><<<gF, 256, 0, stream>>>(
        h_bf, w1_bf, b1, nullptr, nullptr, y1_bf, NT, FF, D, 1.0f);

    // ---- gelu(y1 @ W2 + b2) + x2 -> out (fp32) ----
    gemm_bf16_kernel<1, true, true, false><<<gP, 256, 0, stream>>>(
        y1_bf, w2_bf, b2, x2_f, out, nullptr, NT, D, FF, 1.0f);
}
